// PatchPartitioningModule_12438225289897
// MI455X (gfx1250) — matrix operand order repaired and checked by reading, not compiled, not run on hardware
//
#include <hip/hip_runtime.h>
#include <hip/hip_bf16.h>
#include <math.h>

// ---------------------------------------------------------------------------
// CDNA5 / gfx1250 implementation.
// Convs: LDS-staged im2col + implicit GEMM on v_wmma_f32_16x16x32_f16
// (wave32). Activations/weights fp16, accumulate fp32.
// Workgroup = 256 threads (8 waves); each block does 128 pixels of one output
// row x all output channels. A-fragments come from LDS as 2x ds_load_b128.
// ---------------------------------------------------------------------------

typedef __attribute__((ext_vector_type(16))) _Float16 v16h;
typedef __attribute__((ext_vector_type(8)))  _Float16 v8h;
typedef __attribute__((ext_vector_type(8)))  float    v8f;

#define BATCH 4
#define PATCH_N 256   // (512/32)^2
#define PATCH_P 64

// ---------------------------------------------------------------------------
// fp32 -> fp16 convert
// ---------------------------------------------------------------------------
__global__ __launch_bounds__(256) void f32_to_f16_kernel(const float* __restrict__ X,
                                                         _Float16* __restrict__ Y, int n) {
    int i = blockIdx.x * blockDim.x + threadIdx.x;
    if (i < n) Y[i] = (_Float16)X[i];
}

// ---------------------------------------------------------------------------
// Pack OIHW fp32 weights into [CoutPad][Kpad] fp16 (K = ci*9 + ky*3 + kx),
// zero padded so WMMA runs full K chunks / N tiles.
// ---------------------------------------------------------------------------
__global__ __launch_bounds__(256) void pack_weights_kernel(const float* __restrict__ W,
                                                           _Float16* __restrict__ Wp,
                                                           int Cin, int Cout, int CoutPad,
                                                           int Kpad) {
    int idx = blockIdx.x * blockDim.x + threadIdx.x;
    int total = CoutPad * Kpad;
    if (idx >= total) return;
    int n = idx / Kpad;
    int k = idx - n * Kpad;
    float v = 0.0f;
    if (n < Cout && k < Cin * 9) v = W[n * Cin * 9 + k];
    Wp[idx] = (_Float16)v;
}

// ---------------------------------------------------------------------------
// 3x3 conv (pad 1): LDS im2col stage + WMMA GEMM + coalesced writeout.
// NT = CoutPad/16 (compile-time). One wave handles 16 pixels x CoutPad.
// ---------------------------------------------------------------------------
template <int NT>
__global__ __launch_bounds__(256) void conv3x3_wmma_kernel(
    const _Float16* __restrict__ X, const _Float16* __restrict__ Wp,
    const float* __restrict__ bias, _Float16* __restrict__ Y,
    int Cin, int Cout, int H, int W, int Kpad, int relu) {

    extern __shared__ _Float16 lds[];

    const int CoutPad = NT * 16;
    const int KS = Kpad + 8;          // LDS im2col row stride (keeps 16B align)
    const int OS = 136;               // LDS output row stride (bank-friendly)

    const int lane   = threadIdx.x & 31;
    const int laneHi = lane >> 4;     // 0: lanes 0-15, 1: lanes 16-31
    const int laneLo = lane & 15;
    const int wave   = threadIdx.x >> 5;

    // block -> (b, y, x0): 128-pixel segment of one row
    const int segs  = W >> 7;
    const int x0    = (blockIdx.x % segs) << 7;
    const int rowId = blockIdx.x / segs;       // b*H + y
    const int y     = rowId % H;
    const int b     = rowId / H;
    const int K9    = Cin * 9;

    // ---- stage im2col tile: lds[pix*KS + k], coalesced over pix ----
    const int total = Kpad << 7;               // Kpad * 128
    for (int t = threadIdx.x; t < total; t += 256) {
        const int pix = t & 127;
        const int k   = t >> 7;
        _Float16 v = (_Float16)0.0f;
        if (k < K9) {
            const int ci = k / 9;
            const int r9 = k - ci * 9;
            const int ky = r9 / 3;
            const int kx = r9 - ky * 3;
            const int yy = y + ky - 1;
            const int xx = x0 + pix + kx - 1;
            if (yy >= 0 && yy < H && xx >= 0 && xx < W)
                v = X[(((long)b * Cin + ci) * H + yy) * W + xx];
        }
        lds[pix * KS + k] = v;
    }
    __syncthreads();

    // ---- GEMM: A from LDS (2x b128/chunk), B from packed weights ----
    const int xBase = wave << 4;               // this wave's 16 pixels
    const _Float16* arow = lds + (xBase + laneLo) * KS + laneHi * 16;

    v8f acc[NT];
#pragma unroll
    for (int nt = 0; nt < NT; ++nt) acc[nt] = (v8f){};

    const int nChunks = Kpad >> 5;
    for (int kc = 0; kc < nChunks; ++kc) {
        const int kof = kc << 5;
        // A layout: e0..15 -> K = kof + laneHi*16 + e
        const v8h a0 = *(const v8h*)(arow + kof);
        const v8h a1 = *(const v8h*)(arow + kof + 8);
        v16h a;
#pragma unroll
        for (int e = 0; e < 8; ++e) { a[e] = a0[e]; a[8 + e] = a1[e]; }
#pragma unroll
        for (int nt = 0; nt < NT; ++nt) {
            // B layout: lane half picks K offset 0/16, laneLo picks column
            const v16h bf = *(const v16h*)(Wp + ((long)(nt * 16 + laneLo)) * Kpad
                                           + kof + laneHi * 16);
            acc[nt] = __builtin_amdgcn_wmma_f32_16x16x32_f16(
                false, a, false, bf, (short)0, acc[nt], false, false);
        }
    }
    __syncthreads();   // done reading the im2col stage

    // ---- bias + relu, C tiles -> LDS out[n*OS + pix] (b128 per lane) ----
#pragma unroll
    for (int nt = 0; nt < NT; ++nt) {
        const int n = nt * 16 + laneLo;
        const float bv = (n < Cout) ? bias[n] : 0.0f;
        v8h o;
#pragma unroll
        for (int r = 0; r < 8; ++r) {
            float f = acc[nt][r] + bv;      // row M = laneHi*8 + r
            if (relu) f = fmaxf(f, 0.0f);
            o[r] = (_Float16)f;
        }
        *(v8h*)(lds + n * OS + xBase + laneHi * 8) = o;
    }
    __syncthreads();

    // ---- cooperative coalesced writeout: 16B chunks per channel row ----
    const int chunks = CoutPad * 16;           // 128 px = 16 chunks of 8 fp16
    for (int c = threadIdx.x; c < chunks; c += 256) {
        const int n  = c >> 4;
        const int xo = (c & 15) << 3;
        const v8h o = *(const v8h*)(lds + n * OS + xo);
        *(v8h*)(Y + (((long)b * CoutPad + n) * H + y) * W + x0 + xo) = o;
    }
}

// ---------------------------------------------------------------------------
// MaxPool2d(2) on fp16 NCHW
// ---------------------------------------------------------------------------
__global__ __launch_bounds__(256) void pool2_kernel(const _Float16* __restrict__ X,
                                                    _Float16* __restrict__ Y,
                                                    int C, int Hi, int Wi) {
    const int Ho = Hi >> 1, Wo = Wi >> 1;
    const int total = BATCH * C * Ho * Wo;
    int idx = blockIdx.x * blockDim.x + threadIdx.x;
    if (idx >= total) return;
    int x = idx % Wo; int t = idx / Wo;
    int y = t % Ho;   t /= Ho;
    int c = t % C;    int b = t / C;
    const _Float16* p = X + (((long)b * C + c) * Hi + 2 * y) * Wi + 2 * x;
    float m = fmaxf(fmaxf((float)p[0], (float)p[1]),
                    fmaxf((float)p[Wi], (float)p[Wi + 1]));
    Y[idx] = (_Float16)m;
}

// ---------------------------------------------------------------------------
// Bilinear x2 upsample, align_corners=False (half-pixel, edge clamp) fp16
// ---------------------------------------------------------------------------
__global__ __launch_bounds__(256) void up2_kernel(const _Float16* __restrict__ X,
                                                  _Float16* __restrict__ Y,
                                                  int C, int Hi, int Wi) {
    const int Ho = Hi << 1, Wo = Wi << 1;
    const int total = BATCH * C * Ho * Wo;
    int idx = blockIdx.x * blockDim.x + threadIdx.x;
    if (idx >= total) return;
    int x = idx % Wo; int t = idx / Wo;
    int y = t % Ho;   t /= Ho;
    int c = t % C;    int b = t / C;
    float sy = (y + 0.5f) * 0.5f - 0.5f;
    float sx = (x + 0.5f) * 0.5f - 0.5f;
    float y0f = floorf(sy), x0f = floorf(sx);
    float wy = sy - y0f,    wx = sx - x0f;
    int y0 = (int)y0f, x0 = (int)x0f;
    int y0c = min(max(y0, 0), Hi - 1), y1c = min(max(y0 + 1, 0), Hi - 1);
    int x0c = min(max(x0, 0), Wi - 1), x1c = min(max(x0 + 1, 0), Wi - 1);
    const _Float16* p = X + ((long)b * C + c) * Hi * Wi;
    float v = (1.f - wy) * ((1.f - wx) * (float)p[y0c * Wi + x0c] + wx * (float)p[y0c * Wi + x1c]) +
              wy *        ((1.f - wx) * (float)p[y1c * Wi + x0c] + wx * (float)p[y1c * Wi + x1c]);
    Y[idx] = (_Float16)v;
}

// ---------------------------------------------------------------------------
// flow = tanh(conv8 out[ch 0,1]); deformed = base_grid + flow * temp
// ---------------------------------------------------------------------------
__global__ __launch_bounds__(256) void flow_deform_kernel(const _Float16* __restrict__ act,
                                                          const float* __restrict__ temp,
                                                          float* __restrict__ flow,
                                                          float* __restrict__ def,
                                                          int H, int W) {
    const int total = BATCH * H * W;
    int idx = blockIdx.x * blockDim.x + threadIdx.x;
    if (idx >= total) return;
    int x = idx % W; int t = idx / W;
    int y = t % H;   int b = t / H;
    float fx = tanhf((float)act[(((long)b * 16 + 0) * H + y) * W + x]);
    float fy = tanhf((float)act[(((long)b * 16 + 1) * H + y) * W + x]);
    flow[(((long)b * 2 + 0) * H + y) * W + x] = fx;
    flow[(((long)b * 2 + 1) * H + y) * W + x] = fy;
    const float tm = temp[0];
    float gx = -1.0f + 2.0f * x / (float)(W - 1);
    float gy = -1.0f + 2.0f * y / (float)(H - 1);
    def[(((long)b * H + y) * W + x) * 2 + 0] = gx + fx * tm;
    def[(((long)b * H + y) * W + x) * 2 + 1] = gy + fy * tm;
}

// ---------------------------------------------------------------------------
// Patch extraction: grid_sample bilinear, zeros padding, align_corners=False.
// centers = deformed[:, ::32, ::32, :]. Output [B, N=256, C=3, 64, 64] fp32.
// ---------------------------------------------------------------------------
__global__ __launch_bounds__(256) void patches_kernel(const float* __restrict__ img,
                                                      const float* __restrict__ def,
                                                      float* __restrict__ outp,
                                                      int H, int W) {
    const int total = BATCH * PATCH_N * PATCH_P * PATCH_P;
    int idx = blockIdx.x * blockDim.x + threadIdx.x;
    if (idx >= total) return;
    int j = idx % PATCH_P; int t = idx / PATCH_P;
    int i = t % PATCH_P;   t /= PATCH_P;
    int n = t % PATCH_N;   int b = t / PATCH_N;

    const int cyi = (n >> 4) * 32;
    const int cxi = (n & 15) * 32;
    const float* dc = def + (((long)b * H + cyi) * W + cxi) * 2;
    const float cx = dc[0], cy = dc[1];

    const float lj = -1.0f + 2.0f * j / (float)(PATCH_P - 1);
    const float li = -1.0f + 2.0f * i / (float)(PATCH_P - 1);
    const float gx = cx + lj * ((float)PATCH_P / (float)W);
    const float gy = cy + li * ((float)PATCH_P / (float)H);

    const float ix = ((gx + 1.0f) * W - 1.0f) * 0.5f;
    const float iy = ((gy + 1.0f) * H - 1.0f) * 0.5f;
    const float x0f = floorf(ix), y0f = floorf(iy);
    const float wx1 = ix - x0f,   wy1 = iy - y0f;
    const int x0 = (int)x0f, y0 = (int)y0f;

    const int xs[2] = {x0, x0 + 1};
    const int ys[2] = {y0, y0 + 1};
    const float wxs[2] = {1.0f - wx1, wx1};
    const float wys[2] = {1.0f - wy1, wy1};

#pragma unroll
    for (int c = 0; c < 3; ++c) {
        const float* ip = img + ((long)b * 3 + c) * H * W;
        float acc = 0.0f;
#pragma unroll
        for (int cy2 = 0; cy2 < 2; ++cy2) {
#pragma unroll
            for (int cx2 = 0; cx2 < 2; ++cx2) {
                const int xi = xs[cx2], yi = ys[cy2];
                const float wgt = wxs[cx2] * wys[cy2];
                const bool valid = (xi >= 0) && (xi <= W - 1) && (yi >= 0) && (yi <= H - 1);
                const int xc = min(max(xi, 0), W - 1);
                const int yc = min(max(yi, 0), H - 1);
                acc += ip[yc * W + xc] * (valid ? wgt : 0.0f);
            }
        }
        outp[((((long)b * PATCH_N + n) * 3 + c) * PATCH_P + i) * PATCH_P + j] = acc;
    }
}

// ---------------------------------------------------------------------------
// Host-side orchestration
// ---------------------------------------------------------------------------
extern "C" void kernel_launch(void* const* d_in, const int* in_sizes, int n_in,
                              void* d_out, int out_size, void* d_ws, size_t ws_size,
                              hipStream_t stream) {
    (void)in_sizes; (void)n_in; (void)out_size; (void)ws_size;

    const float* x = (const float*)d_in[0];
    const float* w[8];
    const float* bia[8];
    for (int i = 0; i < 8; ++i) {
        w[i]   = (const float*)d_in[1 + 2 * i];
        bia[i] = (const float*)d_in[2 + 2 * i];
    }
    const float* temp = (const float*)d_in[17];
    float* out = (float*)d_out;

    float* flowOut = out + (long)BATCH * PATCH_N * 3 * PATCH_P * PATCH_P;   // +12,582,912
    float* defOut  = flowOut + (long)BATCH * 2 * 512 * 512;                  // +2,097,152

    // workspace layout
    char* ws = (char*)d_ws;
    _Float16* xh   = (_Float16*)ws;                               // 6 MB
    _Float16* actA = (_Float16*)(ws + ((size_t)8  << 20));        // 64 MB
    _Float16* actB = (_Float16*)(ws + ((size_t)72 << 20));        // 64 MB

    static const int CinArr[8]  = {3, 32, 32, 64, 64, 64, 32, 32};
    static const int CoutArr[8] = {32, 32, 64, 64, 64, 32, 32, 2};
    int Kpad[8], CoutPad[8];
    _Float16* wp[8];
    size_t off = (size_t)136 << 20;
    for (int i = 0; i < 8; ++i) {
        int K = CinArr[i] * 9;
        Kpad[i]    = ((K + 31) / 32) * 32;
        CoutPad[i] = ((CoutArr[i] + 15) / 16) * 16;
        wp[i] = (_Float16*)(ws + off);
        off += ((size_t)CoutPad[i] * Kpad[i] * sizeof(_Float16) + 255) & ~(size_t)255;
    }

    // pack weights
    for (int i = 0; i < 8; ++i) {
        int total = CoutPad[i] * Kpad[i];
        pack_weights_kernel<<<(total + 255) / 256, 256, 0, stream>>>(
            w[i], wp[i], CinArr[i], CoutArr[i], CoutPad[i], Kpad[i]);
    }
    // convert input to fp16
    {
        int n = BATCH * 3 * 512 * 512;
        f32_to_f16_kernel<<<(n + 255) / 256, 256, 0, stream>>>(x, xh, n);
    }

    auto conv = [&](const _Float16* X, int li, _Float16* Y, int H, int W, int relu) {
        const unsigned blocks = (unsigned)((long)BATCH * H * (W / 128));
        size_t stageE = (size_t)128 * (Kpad[li] + 8);
        size_t outE   = (size_t)CoutPad[li] * 136;
        size_t sh     = 2 * (stageE > outE ? stageE : outE);
        switch (CoutPad[li] / 16) {
        case 1:
            conv3x3_wmma_kernel<1><<<blocks, 256, sh, stream>>>(
                X, wp[li], bia[li], Y, CinArr[li], CoutArr[li], H, W, Kpad[li], relu);
            break;
        case 2:
            conv3x3_wmma_kernel<2><<<blocks, 256, sh, stream>>>(
                X, wp[li], bia[li], Y, CinArr[li], CoutArr[li], H, W, Kpad[li], relu);
            break;
        default:
            conv3x3_wmma_kernel<4><<<blocks, 256, sh, stream>>>(
                X, wp[li], bia[li], Y, CinArr[li], CoutArr[li], H, W, Kpad[li], relu);
            break;
        }
    };
    auto pool = [&](const _Float16* X, _Float16* Y, int C, int Hi, int Wi) {
        int total = BATCH * C * (Hi / 2) * (Wi / 2);
        pool2_kernel<<<(total + 255) / 256, 256, 0, stream>>>(X, Y, C, Hi, Wi);
    };
    auto up = [&](const _Float16* X, _Float16* Y, int C, int Hi, int Wi) {
        int total = BATCH * C * (Hi * 2) * (Wi * 2);
        up2_kernel<<<(total + 255) / 256, 256, 0, stream>>>(X, Y, C, Hi, Wi);
    };

    // flow net
    conv(xh,   0, actA, 512, 512, 1);   // 3  -> 32 @512
    conv(actA, 1, actB, 512, 512, 1);   // 32 -> 32 @512
    pool(actB, actA, 32, 512, 512);     // -> 32 @256
    conv(actA, 2, actB, 256, 256, 1);   // 32 -> 64 @256
    conv(actB, 3, actA, 256, 256, 1);   // 64 -> 64 @256
    pool(actA, actB, 64, 256, 256);     // -> 64 @128
    up(actB,  actA, 64, 128, 128);      // -> 64 @256
    conv(actA, 4, actB, 256, 256, 1);   // 64 -> 64 @256
    conv(actB, 5, actA, 256, 256, 1);   // 64 -> 32 @256
    up(actA,  actB, 32, 256, 256);      // -> 32 @512
    conv(actB, 6, actA, 512, 512, 1);   // 32 -> 32 @512
    conv(actA, 7, actB, 512, 512, 0);   // 32 -> 2(pad16) @512, linear

    // tanh -> flow + deformed grid
    {
        int total = BATCH * 512 * 512;
        flow_deform_kernel<<<(total + 255) / 256, 256, 0, stream>>>(
            actB, temp, flowOut, defOut, 512, 512);
    }
    // patch extraction (grid_sample)
    {
        long total = (long)BATCH * PATCH_N * PATCH_P * PATCH_P;
        patches_kernel<<<(unsigned)((total + 255) / 256), 256, 0, stream>>>(
            x, defOut, out, 512, 512);
    }
}
